// MultiHeadAttentionAdv_16698832847502
// MI455X (gfx1250) — compile-verified, matrix operand order checked by reading, not run on hardware
//
#include <hip/hip_runtime.h>

#define HH      16
#define DKK     64
#define INDIM   768
#define DMODEL  1024
#define BB      16
#define SS      1024

typedef __attribute__((ext_vector_type(16))) __bf16 bf16x16;
typedef __attribute__((ext_vector_type(8)))  float  f32x8;

union FragAB {
    bf16x16 v;
    uint4   q[2];
};

__device__ __forceinline__ unsigned short f2bf(float f) {
    union { float f; unsigned u; } v; v.f = f;
    unsigned r = v.u + 0x7FFFu + ((v.u >> 16) & 1u);   // round-to-nearest-even
    return (unsigned short)(r >> 16);
}
__device__ __forceinline__ unsigned pk2(float lo, float hi) {
    return (unsigned)f2bf(lo) | ((unsigned)f2bf(hi) << 16);
}

#define WMMA_BF16(A, Bf, C) \
    __builtin_amdgcn_wmma_f32_16x16x32_bf16(false, (A), false, (Bf), (short)0, (C), false, false)

// ---------------------------------------------------------------------------
// Kernel 1a: transpose+convert f32 [K][N] weights -> bf16 [N][K] (n-major)
// ---------------------------------------------------------------------------
__global__ void pack_wt(const float* __restrict__ W, unsigned short* __restrict__ Wt,
                        int K, int N) {
    int idx = blockIdx.x * blockDim.x + threadIdx.x;
    if (idx >= K * N) return;
    int k = idx / N, n = idx - k * N;
    Wt[(size_t)n * K + k] = f2bf(W[idx]);
}

// ---------------------------------------------------------------------------
// Kernel 1b: convert f32 activations -> bf16, 8 elements/thread, vectorized.
// grid.z selects query/key/value.
// ---------------------------------------------------------------------------
__global__ __launch_bounds__(256) void pack_in(
    const float* __restrict__ q_in, const float* __restrict__ k_in,
    const float* __restrict__ v_in,
    unsigned short* __restrict__ qb, unsigned short* __restrict__ kb,
    unsigned short* __restrict__ vb)
{
    const int z = blockIdx.z;
    const float*    src = z == 0 ? q_in : (z == 1 ? k_in : v_in);
    unsigned short* dst = z == 0 ? qb   : (z == 1 ? kb   : vb);
    size_t i = ((size_t)blockIdx.x * blockDim.x + threadIdx.x) * 8;
    float4 x = *(const float4*)(src + i);
    float4 y = *(const float4*)(src + i + 4);
    *(uint4*)(dst + i) = make_uint4(pk2(x.x, x.y), pk2(x.z, x.w),
                                    pk2(y.x, y.y), pk2(y.z, y.w));
}

// ---------------------------------------------------------------------------
// Kernel 2: fused QKV projection.  mode = blockIdx.z (0:Q, 1:K, 2:V-transposed)
// M = B*S = 16384, N = 1024, K = 768.  Per wave: 32x64 tile (2x4 C-frags).
// ---------------------------------------------------------------------------
__global__ __launch_bounds__(256) void qkv_proj(
    const unsigned short* __restrict__ qib, const unsigned short* __restrict__ kib,
    const unsigned short* __restrict__ vib,
    const unsigned short* __restrict__ Wqt, const unsigned short* __restrict__ Wkt,
    const unsigned short* __restrict__ Wvt,
    const float* __restrict__ bq, const float* __restrict__ bk, const float* __restrict__ bv,
    unsigned short* __restrict__ Qb, unsigned short* __restrict__ Kb,
    unsigned short* __restrict__ Vtb)
{
    const int mode = blockIdx.z;
    const unsigned short* src  = mode == 0 ? qib : (mode == 1 ? kib : vib);
    const unsigned short* Wt   = mode == 0 ? Wqt : (mode == 1 ? Wkt : Wvt);
    const float*          bias = mode == 0 ? bq  : (mode == 1 ? bk  : bv);

    const int lane   = threadIdx.x & 31;
    const int wave   = threadIdx.x >> 5;
    const int col    = lane & 15;           // A: row m; B: col n
    const int ko     = (lane >> 4) * 8;     // K sub-offset per lane half
    const int rowoff = ko;                  // C: row offset per lane half
    const int m0 = blockIdx.x * 256 + wave * 32;
    const int n0 = blockIdx.y * 64;

    f32x8 cacc[2][4] = {};
    const unsigned short* ap0 = src + (size_t)(m0 + col) * INDIM + ko;
    const unsigned short* ap1 = ap0 + (size_t)16 * INDIM;

    for (int k0 = 0; k0 < INDIM; k0 += 32) {
        FragAB a0, a1;
        a0.q[0] = *(const uint4*)(ap0 + k0);
        a0.q[1] = *(const uint4*)(ap0 + k0 + 16);
        a1.q[0] = *(const uint4*)(ap1 + k0);
        a1.q[1] = *(const uint4*)(ap1 + k0 + 16);
        #pragma unroll
        for (int t = 0; t < 4; ++t) {
            const unsigned short* bp = Wt + (size_t)(n0 + t * 16 + col) * INDIM + k0 + ko;
            __builtin_prefetch(bp + 32, 0, 1);   // global_prefetch_b8 next K tile
            FragAB bfr;
            bfr.q[0] = *(const uint4*)bp;
            bfr.q[1] = *(const uint4*)(bp + 16);
            cacc[0][t] = WMMA_BF16(a0.v, bfr.v, cacc[0][t]);
            cacc[1][t] = WMMA_BF16(a1.v, bfr.v, cacc[1][t]);
        }
    }

    #pragma unroll
    for (int g = 0; g < 2; ++g) {
        #pragma unroll
        for (int t = 0; t < 4; ++t) {
            const int n = n0 + t * 16 + col;
            const float badd = bias[n];
            const int h = n >> 6, d = n & 63;
            #pragma unroll
            for (int j = 0; j < 8; ++j) {
                const int m = m0 + g * 16 + rowoff + j;
                const int bidx = m >> 10, s = m & (SS - 1);
                const float val = cacc[g][t][j] + badd;
                if (mode == 2)
                    Vtb[(((size_t)bidx * HH + h) * DKK + d) * SS + s] = f2bf(val);
                else if (mode == 0)
                    Qb[(((size_t)bidx * HH + h) * SS + s) * DKK + d] = f2bf(val);
                else
                    Kb[(((size_t)bidx * HH + h) * SS + s) * DKK + d] = f2bf(val);
            }
        }
    }
}

// ---------------------------------------------------------------------------
// Kernel 3: attention. grid = (S/128, B*H), 8 waves; each wave: 16 q-rows.
// Pass 1: stream keys, exp*mask, rowsum + x += P@V (P re-fragmented via LDS).
// Pass 2: recompute scores, write normalized p_attn once (nontemporal).
// ---------------------------------------------------------------------------
__global__ __launch_bounds__(256) void attn(
    const unsigned short* __restrict__ Qb, const unsigned short* __restrict__ Kb,
    const unsigned short* __restrict__ Vtb, const int* __restrict__ mask,
    unsigned short* __restrict__ Xb, float* __restrict__ pattn)
{
    __shared__ unsigned short lds[8 * 16 * 32];   // 8 KB: per-wave 16x32 bf16 P tile

    const int lane   = threadIdx.x & 31;
    const int wave   = threadIdx.x >> 5;
    const int col    = lane & 15;
    const int ko     = (lane >> 4) * 8;
    const int rowoff = ko;
    const int bh = blockIdx.y;
    const int b  = bh >> 4;
    const int h  = bh & 15;
    const int q0 = blockIdx.x * 128 + wave * 16;

    const unsigned short* Qp = Qb  + (size_t)bh * SS * DKK;
    const unsigned short* Kp = Kb  + (size_t)bh * SS * DKK;
    const unsigned short* Vp = Vtb + (size_t)bh * DKK * SS;
    unsigned short* ldsw = lds + wave * (16 * 32);

    // Q fragments for k0=0 and k0=32 (held in registers for the whole kernel)
    FragAB aq0, aq1;
    {
        const unsigned short* p = Qp + (size_t)(q0 + col) * DKK + ko;
        aq0.q[0] = *(const uint4*)p;        aq0.q[1] = *(const uint4*)(p + 16);
        aq1.q[0] = *(const uint4*)(p + 32); aq1.q[1] = *(const uint4*)(p + 48);
    }

    float maskv[8], rsum[8];
    #pragma unroll
    for (int j = 0; j < 8; ++j) {
        maskv[j] = (float)mask[b * SS + q0 + rowoff + j];
        rsum[j]  = 0.f;
    }
    f32x8 xacc[4] = {};

    for (int kt2 = 0; kt2 < SS / 32; ++kt2) {
        #pragma unroll
        for (int t = 0; t < 2; ++t) {
            const int kt = kt2 * 2 + t;
            const unsigned short* kp = Kp + (size_t)(kt * 16 + col) * DKK + ko;
            FragAB b0, b1;
            b0.q[0] = *(const uint4*)kp;        b0.q[1] = *(const uint4*)(kp + 16);
            b1.q[0] = *(const uint4*)(kp + 32); b1.q[1] = *(const uint4*)(kp + 48);
            f32x8 c = {};
            c = WMMA_BF16(aq0.v, b0.v, c);
            c = WMMA_BF16(aq1.v, b1.v, c);
            #pragma unroll
            for (int j = 0; j < 8; ++j) {
                const float e = __expf(c[j] * 0.125f) * maskv[j];
                rsum[j] += e;
                ldsw[(rowoff + j) * 32 + t * 16 + col] = f2bf(e);
            }
        }
        asm volatile("s_wait_dscnt 0" ::: "memory");   // LDS RAW: C-layout -> A-layout
        FragAB pa;
        {
            const unsigned short* p = ldsw + col * 32 + ko;
            pa.q[0] = *(const uint4*)p;
            pa.q[1] = *(const uint4*)(p + 16);
        }
        #pragma unroll
        for (int d = 0; d < 4; ++d) {
            const unsigned short* vp = Vp + (size_t)(d * 16 + col) * SS + kt2 * 32 + ko;
            __builtin_prefetch(vp + 32, 0, 1);
            FragAB bv;
            bv.q[0] = *(const uint4*)vp;
            bv.q[1] = *(const uint4*)(vp + 16);
            xacc[d] = WMMA_BF16(pa.v, bv.v, xacc[d]);
        }
    }

    // rowsum reduction across the 16 lanes of each half (wave32 xor shuffles)
    float inv[8], minv[8];
    #pragma unroll
    for (int j = 0; j < 8; ++j) {
        float s = rsum[j];
        s += __shfl_xor(s, 1, 32);
        s += __shfl_xor(s, 2, 32);
        s += __shfl_xor(s, 4, 32);
        s += __shfl_xor(s, 8, 32);
        inv[j]  = 1.0f / (s + 1e-8f);
        minv[j] = maskv[j] * inv[j];
    }

    // store normalized x as bf16 [B*S, D_MODEL] (A operand for out_proj)
    #pragma unroll
    for (int d = 0; d < 4; ++d) {
        #pragma unroll
        for (int j = 0; j < 8; ++j) {
            const int row = q0 + rowoff + j;
            const int n   = h * 64 + d * 16 + col;
            Xb[((size_t)b * SS + row) * DMODEL + n] = f2bf(xacc[d][j] * inv[j]);
        }
    }

    // pass 2: recompute scores (K tiles are cache-hot), write p_attn once,
    // nontemporal (1 GB stream, never re-read -> keep it out of L2)
    float* pa_out = pattn + (size_t)bh * SS * SS;
    for (int kt = 0; kt < SS / 16; ++kt) {
        const unsigned short* kp = Kp + (size_t)(kt * 16 + col) * DKK + ko;
        FragAB b0, b1;
        b0.q[0] = *(const uint4*)kp;        b0.q[1] = *(const uint4*)(kp + 16);
        b1.q[0] = *(const uint4*)(kp + 32); b1.q[1] = *(const uint4*)(kp + 48);
        f32x8 c = {};
        c = WMMA_BF16(aq0.v, b0.v, c);
        c = WMMA_BF16(aq1.v, b1.v, c);
        #pragma unroll
        for (int j = 0; j < 8; ++j) {
            const float e = __expf(c[j] * 0.125f) * minv[j];
            __builtin_nontemporal_store(
                e, pa_out + (size_t)(q0 + rowoff + j) * SS + kt * 16 + col);
        }
    }
}

// ---------------------------------------------------------------------------
// Kernel 4: out = X @ Wo + bo.  M=16384, N=1024, K=1024. Per wave 32x64 tile.
// ---------------------------------------------------------------------------
__global__ __launch_bounds__(256) void out_proj(
    const unsigned short* __restrict__ Xb, const unsigned short* __restrict__ Wot,
    const float* __restrict__ bo, float* __restrict__ out)
{
    const int lane   = threadIdx.x & 31;
    const int wave   = threadIdx.x >> 5;
    const int col    = lane & 15;
    const int ko     = (lane >> 4) * 8;
    const int rowoff = ko;
    const int m0 = blockIdx.x * 256 + wave * 32;
    const int n0 = blockIdx.y * 64;

    f32x8 c[2][4] = {};
    const unsigned short* ap0 = Xb  + (size_t)(m0 + col) * DMODEL + ko;
    const unsigned short* ap1 = ap0 + (size_t)16 * DMODEL;
    const unsigned short* bp0 = Wot + (size_t)(n0 + col) * DMODEL + ko;

    for (int k0 = 0; k0 < DMODEL; k0 += 32) {
        FragAB a0, a1;
        a0.q[0] = *(const uint4*)(ap0 + k0);
        a0.q[1] = *(const uint4*)(ap0 + k0 + 16);
        a1.q[0] = *(const uint4*)(ap1 + k0);
        a1.q[1] = *(const uint4*)(ap1 + k0 + 16);
        #pragma unroll
        for (int t = 0; t < 4; ++t) {
            const unsigned short* bp = bp0 + (size_t)t * 16 * DMODEL + k0;
            __builtin_prefetch(bp + 32, 0, 1);
            FragAB bfr;
            bfr.q[0] = *(const uint4*)bp;
            bfr.q[1] = *(const uint4*)(bp + 16);
            c[0][t] = WMMA_BF16(a0.v, bfr.v, c[0][t]);
            c[1][t] = WMMA_BF16(a1.v, bfr.v, c[1][t]);
        }
    }

    #pragma unroll
    for (int g = 0; g < 2; ++g) {
        #pragma unroll
        for (int t = 0; t < 4; ++t) {
            const float badd = bo[n0 + t * 16 + col];
            #pragma unroll
            for (int j = 0; j < 8; ++j)
                __builtin_nontemporal_store(
                    c[g][t][j] + badd,
                    out + (size_t)(m0 + g * 16 + rowoff + j) * DMODEL + n0 + t * 16 + col);
        }
    }
}

// ---------------------------------------------------------------------------
extern "C" void kernel_launch(void* const* d_in, const int* in_sizes, int n_in,
                              void* d_out, int out_size, void* d_ws, size_t ws_size,
                              hipStream_t stream) {
    const float* q_in = (const float*)d_in[0];
    const float* k_in = (const float*)d_in[1];
    const float* v_in = (const float*)d_in[2];
    const int*   mask = (const int*)d_in[3];
    const float* Wq = (const float*)d_in[4];  const float* bq = (const float*)d_in[5];
    const float* Wk = (const float*)d_in[6];  const float* bk = (const float*)d_in[7];
    const float* Wv = (const float*)d_in[8];  const float* bv = (const float*)d_in[9];
    const float* Wo = (const float*)d_in[10]; const float* bo = (const float*)d_in[11];

    // workspace carve-out (all bf16 staging buffers), 256B-aligned
    char* w = (char*)d_ws;
    auto carve = [&](size_t bytes) -> char* {
        char* p = w;
        w += (bytes + 255) & ~(size_t)255;
        return p;
    };
    unsigned short* Wqt = (unsigned short*)carve((size_t)INDIM * DMODEL * 2);
    unsigned short* Wkt = (unsigned short*)carve((size_t)INDIM * DMODEL * 2);
    unsigned short* Wvt = (unsigned short*)carve((size_t)INDIM * DMODEL * 2);
    unsigned short* Wot = (unsigned short*)carve((size_t)DMODEL * DMODEL * 2);
    unsigned short* Qib = (unsigned short*)carve((size_t)BB * SS * INDIM * 2);
    unsigned short* Kib = (unsigned short*)carve((size_t)BB * SS * INDIM * 2);
    unsigned short* Vib = (unsigned short*)carve((size_t)BB * SS * INDIM * 2);
    unsigned short* Qb  = (unsigned short*)carve((size_t)BB * HH * SS * DKK * 2);
    unsigned short* Kb  = (unsigned short*)carve((size_t)BB * HH * SS * DKK * 2);
    unsigned short* Vtb = (unsigned short*)carve((size_t)BB * HH * DKK * SS * 2);
    unsigned short* Xb  = (unsigned short*)carve((size_t)BB * SS * DMODEL * 2);

    float* out   = (float*)d_out;
    float* pattn = out + (size_t)BB * SS * DMODEL;   // p_attn follows out

    pack_wt<<<(INDIM * DMODEL + 255) / 256, 256, 0, stream>>>(Wq, Wqt, INDIM, DMODEL);
    pack_wt<<<(INDIM * DMODEL + 255) / 256, 256, 0, stream>>>(Wk, Wkt, INDIM, DMODEL);
    pack_wt<<<(INDIM * DMODEL + 255) / 256, 256, 0, stream>>>(Wv, Wvt, INDIM, DMODEL);
    pack_wt<<<(DMODEL * DMODEL + 255) / 256, 256, 0, stream>>>(Wo, Wot, DMODEL, DMODEL);

    pack_in<<<dim3((size_t)BB * SS * INDIM / 8 / 256, 1, 3), 256, 0, stream>>>(
        q_in, k_in, v_in, Qib, Kib, Vib);

    qkv_proj<<<dim3(BB * SS / 256, DMODEL / 64, 3), 256, 0, stream>>>(
        Qib, Kib, Vib, Wqt, Wkt, Wvt, bq, bk, bv, Qb, Kb, Vtb);

    attn<<<dim3(SS / 128, BB * HH), 256, 0, stream>>>(Qb, Kb, Vtb, mask, Xb, pattn);

    out_proj<<<dim3(BB * SS / 256, DMODEL / 64), 256, 0, stream>>>(Xb, Wot, bo, out);
}
